// DiscriminatorNet_2972117369002
// MI455X (gfx1250) — matrix operand order repaired and checked by reading, not compiled, not run on hardware
//
#include <hip/hip_runtime.h>

typedef __attribute__((ext_vector_type(16))) _Float16 v16h;
typedef __attribute__((ext_vector_type(8)))  _Float16 v8h;   // 16B aligned
typedef __attribute__((ext_vector_type(8)))  float    v8f;
typedef unsigned int v4u __attribute__((ext_vector_type(4)));
typedef int          v8i __attribute__((ext_vector_type(8)));
typedef int          v4i __attribute__((ext_vector_type(4)));

#define LRELU_SLOPE 0.2f
#define IN_EPS      1e-5f

#if defined(__gfx1250__) && __has_builtin(__builtin_amdgcn_tensor_load_to_lds) && \
    __has_builtin(__builtin_amdgcn_s_wait_tensorcnt)
#define ATHENA_USE_TDM 1
#else
#define ATHENA_USE_TDM 0
#endif

// ---------------------------------------------------------------- zero fill
__global__ void zero_kernel(float* __restrict__ p, int n) {
    int i = blockIdx.x * blockDim.x + threadIdx.x;
    if (i < n) p[i] = 0.0f;
}

// ---------------------------------------------------------------- weight convert
// f32 [M,K] -> f16 [Mp,Kp], zero padded, K permuted inside each 32-block so a
// lane's A fragment (ISA 7.12.2 16-bit A layout) is a contiguous 16-half slice:
// block-of-8 order [0,2,1,3] (an involution).
__global__ void wcvt_kernel(const float* __restrict__ W, _Float16* __restrict__ Wh,
                            int M, int K, int Mp, int Kp) {
    int i = blockIdx.x * blockDim.x + threadIdx.x;
    if (i >= Mp * Kp) return;
    int m = i / Kp, kpos = i % Kp;
    int j = kpos & 31, base = kpos & ~31;
    int b = j >> 3, nb = ((b & 1) << 1) | (b >> 1);
    int k = base + nb * 8 + (j & 7);
    float v = (m < M && k < K) ? W[(size_t)m * K + k] : 0.0f;
    Wh[(size_t)m * Kp + kpos] = (_Float16)v;
}

// ---------------------------------------------------------------- fragment load
__device__ __forceinline__ v16h ldfrag(const _Float16* p) {
    v8h lo = *(const v8h*)p;          // ds_load_b128
    v8h hi = *(const v8h*)(p + 8);    // ds_load_b128
    return __builtin_shufflevector(lo, hi, 0, 1, 2, 3, 4, 5, 6, 7,
                                   8, 9, 10, 11, 12, 13, 14, 15);
}

// ---------------------------------------------------------------- fused WMMA GEMM
// Y = Wh[Mp,Kp](f16, fragment-permuted) x B[K,N] with B generated on the fly.
//   MODE 0: adjacency conv (edge gather -> GEMM -> scatter-add both endpoints)
//   MODE 1: circular-padded k4 s2 conv (im2col -> GEMM -> store)
// Block: 128 threads (4 wave32s), tile 64(M) x 128(N), K stepped by 32.
// A tile staged to LDS by the Tensor Data Mover when available.
template <int MODE>
__global__ void __launch_bounds__(128)
wmma_stage_kernel(const float*    __restrict__ X,   // [Cin, T] activations (f32)
                  const _Float16* __restrict__ Wt,  // [Mp, Kp] f16 weights
                  const int*      __restrict__ idx, // [D, 2E] (MODE 0)
                  float* __restrict__ Y,
                  int M, int K, int Kp, int Mp, int T,
                  int lgE, int lgL,                 // MODE 0
                  int lgLout, int lgLin)            // MODE 1
{
    __shared__ _Float16 As[64][40];    // [m][k-frag-pos], row stride 80B
    __shared__ _Float16 Bs[128][40];   // [n][k],          row stride 80B
    __shared__ int E0[128], E1[128], DL[128];

    const int tid  = threadIdx.x;
    const int lane = tid & 31;
    const int wave = tid >> 5;
    const int g    = lane >> 4;
    const int ln   = lane & 15;
    const int wm   = (wave >> 1) * 32;   // wave's M offset in 64
    const int wn   = (wave & 1) * 64;    // wave's N offset in 128

    const int nBase = blockIdx.x * 128;
    const int mBase = blockIdx.y * 64;

    // ---- per-column metadata (thread owns column n = tid)
    int e0 = 0, e1 = 0, dl = 0, bpos = 0;
    if (MODE == 0) {
        int n_g = nBase + tid;
        int d   = n_g >> lgE;
        int e   = n_g & ((1 << lgE) - 1);
        const int* ip = idx + ((size_t)d << (lgE + 1));
        e0 = ip[2 * e];
        e1 = ip[2 * e + 1];
        dl = d << lgL;
        E0[tid] = e0; E1[tid] = e1; DL[tid] = dl;
    } else {
        int n_g = nBase + tid;
        int d   = n_g >> lgLout;
        int to  = n_g & ((1 << lgLout) - 1);
        bpos    = (d << lgLin) + 2 * to;
    }
    __syncthreads();

    v8f acc[2][4] = {};

    for (int k0 = 0; k0 < Kp; k0 += 32) {
        // -------- stage A tile [64][32] from pre-permuted f16 weights
#if ATHENA_USE_TDM
        if (tid < 32) {  // wave 0 issues one TDM descriptor (EXEC ignored by TDM)
            unsigned lds_a = (unsigned)(unsigned long long)(uintptr_t)&As[0][0];
            unsigned long long ga =
                (unsigned long long)(uintptr_t)&Wt[(size_t)mBase * Kp + k0];
            v4u g0;
            g0[0] = 1u;                                    // count=1, load, user
            g0[1] = lds_a;                                 // lds_addr
            g0[2] = (unsigned)(ga & 0xffffffffu);          // global_addr lo
            g0[3] = (unsigned)((ga >> 32) & 0x01ffffffu) | (2u << 30); // hi | type=2
            v8i g1;
            unsigned rows = (unsigned)(Mp - mBase);
            g1[0] = (int)((1u << 16)      // data_size = 2 bytes
                        | (1u << 20)      // pad_enable
                        | (3u << 22)      // pad_interval: 16 DWORDs (one 64B row)
                        | (3u << 25));    // pad_amount: 4 DWORDs (16B -> stride 80B)
            g1[1] = (int)(((unsigned)Kp & 0xffffu) << 16);             // dim0 lo
            g1[2] = (int)(((unsigned)Kp >> 16) | ((rows & 0xffffu) << 16)); // dim0 hi|dim1 lo
            g1[3] = (int)((rows >> 16) | (32u << 16));                 // dim1 hi|tile_dim0=32
            g1[4] = (int)64u;                                          // tile_dim1=64
            g1[5] = (int)Kp;                                           // dim0_stride lo
            g1[6] = 0;
            g1[7] = 0;
            v4i gz = {0, 0, 0, 0};
#if __clang_major__ >= 23
            v8i gz8 = {0, 0, 0, 0, 0, 0, 0, 0};
            __builtin_amdgcn_tensor_load_to_lds(g0, g1, gz, gz, gz8, 0);
#else
            __builtin_amdgcn_tensor_load_to_lds(g0, g1, gz, gz, 0);
#endif
            __builtin_amdgcn_s_wait_tensorcnt(0);
        }
#else
        // fallback: plain 16B tile copies
        for (int i = tid; i < 64 * 4; i += 128) {
            int m = i >> 2, c = i & 3;
            *(int4*)&As[m][c * 8] =
                *(const int4*)&Wt[(size_t)(mBase + m) * Kp + k0 + c * 8];
        }
        if (k0 + 32 < Kp && tid < 64)
            __builtin_prefetch(&Wt[(size_t)(mBase + tid) * Kp + k0 + 32], 0, 1);
#endif

        // -------- stage B tile: thread gathers its column n=tid (branchless)
#pragma unroll 8
        for (int j = 0; j < 32; ++j) {
            int gk = k0 + j;
            bool ok = gk < K;
            size_t off;
            if (MODE == 0) {
                int c = gk >> 1;
                int p = (gk & 1) ? e1 : e0;
                off = (size_t)c * T + dl + p;
            } else {
                int c = gk >> 2, r = gk & 3;
                int pos = bpos + r - 1;                     // circular pad by 1
                pos = (pos < 0) ? pos + T : ((pos >= T) ? pos - T : pos);
                off = (size_t)c * T + pos;
            }
            float v = X[ok ? off : 0];
            Bs[tid][((j & 8) << 1) | ((j & 16) >> 1) | (j & 7)] = (_Float16)(ok ? v : 0.0f);
        }
        __syncthreads();

        // -------- fragments (contiguous 32B slices) + 8 WMMAs per wave
        v16h a0 = ldfrag(&As[wm + ln][g * 16]);
        v16h a1 = ldfrag(&As[wm + 16 + ln][g * 16]);
        v16h bf[4];
#pragma unroll
        for (int tn = 0; tn < 4; ++tn)
            bf[tn] = ldfrag(&Bs[wn + tn * 16 + ln][g * 16]);
#pragma unroll
        for (int tn = 0; tn < 4; ++tn) {
            acc[0][tn] = __builtin_amdgcn_wmma_f32_16x16x32_f16(
                false, a0, false, bf[tn], (short)0, acc[0][tn], false, false);
            acc[1][tn] = __builtin_amdgcn_wmma_f32_16x16x32_f16(
                false, a1, false, bf[tn], (short)0, acc[1][tn], false, false);
        }
        __syncthreads();
    }

    // -------- epilogue: C/D layout row = g*8 + e, col = lane%16
    // Hoist per-column metadata (4 columns per thread) and make the row guard
    // branchless (clamp row, zero dead contributions) so the scatter is a
    // straight-line run of global_atomic_add_f32.
    int cOff0[4], cOff1[4], colIx[4];
#pragma unroll
    for (int tn = 0; tn < 4; ++tn) {
        int cl = wn + tn * 16 + ln;
        colIx[tn] = nBase + cl;
        if (MODE == 0) {
            cOff0[tn] = DL[cl] + E0[cl];
            cOff1[tn] = DL[cl] + E1[cl];
        }
    }
#pragma unroll
    for (int tm = 0; tm < 2; ++tm)
#pragma unroll
        for (int e = 0; e < 8; ++e) {
            int row  = mBase + wm + tm * 16 + g * 8 + e;
            bool okr = row < M;
            int rowc = okr ? row : (M - 1);
            if (MODE == 0) {
                float* rY = Y + (size_t)rowc * T;
#pragma unroll
                for (int tn = 0; tn < 4; ++tn) {
                    float v = okr ? acc[tm][tn][e] : 0.0f;
                    atomicAdd(rY + cOff0[tn], v);
                    atomicAdd(rY + cOff1[tn], v);
                }
            } else {
                float* rY = Y + (size_t)rowc * (size_t)(T >> 1);
#pragma unroll
                for (int tn = 0; tn < 4; ++tn) {
                    if (okr) rY[colIx[tn]] = acc[tm][tn][e];
                }
            }
        }
}

// ---------------------------------------------------------------- instance norm + leaky relu
__global__ void __launch_bounds__(256)
norm_lrelu_kernel(float* __restrict__ x, int Nc, int do_norm)
{
    float* p = x + (size_t)blockIdx.x * Nc;
    __shared__ float red[256];
    float mean = 0.0f, inv = 1.0f;
    if (do_norm) {
        float s = 0.0f, s2 = 0.0f;
        for (int i = threadIdx.x; i < Nc; i += 256) {
            float v = p[i]; s += v; s2 += v * v;
        }
        red[threadIdx.x] = s; __syncthreads();
        for (int o = 128; o > 0; o >>= 1) {
            if (threadIdx.x < o) red[threadIdx.x] += red[threadIdx.x + o];
            __syncthreads();
        }
        float sum = red[0]; __syncthreads();
        red[threadIdx.x] = s2; __syncthreads();
        for (int o = 128; o > 0; o >>= 1) {
            if (threadIdx.x < o) red[threadIdx.x] += red[threadIdx.x + o];
            __syncthreads();
        }
        float sum2 = red[0]; __syncthreads();
        mean = sum / (float)Nc;
        float var = sum2 / (float)Nc - mean * mean;
        inv = rsqrtf(var + IN_EPS);
    }
    for (int i = threadIdx.x; i < Nc; i += 256) {
        float v = (p[i] - mean) * inv;
        p[i] = (v > 0.0f) ? v : LRELU_SLOPE * v;
    }
}

// ---------------------------------------------------------------- final 1x1 conv + sigmoid
__global__ void __launch_bounds__(256)
final_kernel(const float* __restrict__ x, const float* __restrict__ w,
             float* __restrict__ out, int C, int Nt)
{
    int t = blockIdx.x * blockDim.x + threadIdx.x;
    if (t >= Nt) return;
    float s = 0.0f;
    for (int c = 0; c < C; ++c) s += w[c] * x[(size_t)c * Nt + t];
    out[t] = 1.0f / (1.0f + __expf(-s));
}

// ---------------------------------------------------------------- host orchestration
static inline int ilog2i(int x) { int r = 0; while ((1 << r) < x) ++r; return r; }
static inline int rup(int x, int a) { return (x + a - 1) / a * a; }

extern "C" void kernel_launch(void* const* d_in, const int* in_sizes, int n_in,
                              void* d_out, int out_size, void* d_ws, size_t ws_size,
                              hipStream_t stream) {
    (void)in_sizes; (void)n_in; (void)out_size; (void)ws_size;

    const float* seq        = (const float*)d_in[0];
    const int*   adj_idx[4] = {(const int*)d_in[1], (const int*)d_in[2],
                               (const int*)d_in[3], (const int*)d_in[4]};
    const float* w_adj[4]   = {(const float*)d_in[5], (const float*)d_in[6],
                               (const float*)d_in[7], (const float*)d_in[8]};
    const float* w_conv[6]  = {(const float*)d_in[9],  (const float*)d_in[10],
                               (const float*)d_in[11], (const float*)d_in[12],
                               (const float*)d_in[13], (const float*)d_in[14]};
    const float* w_final    = (const float*)d_in[15];
    float* out = (float*)d_out;

    const int D = 128;
    const int adjCo[4]  = {32, 64, 128, 256};
    const int convCo[6] = {64, 128, 256, 512, 1024, 2048};
    const int Ls[4]     = {512, 256, 128, 64};
    const int Es[4]     = {4096, 2048, 1024, 512};

    const size_t BUF = 2u * 1024u * 1024u + 256u;  // floats per activation buffer
    float* bAdj = (float*)d_ws;
    float* bC1  = bAdj + BUF;
    float* bC2  = bC1 + BUF;
    _Float16* whBase = (_Float16*)(bC2 + BUF);

    // ---- per-launch weight conversion into padded/permuted f16
    // slots 0..3: adjacency, 4..9: conv
    const float* wsrc[10];
    int wM[10], wK[10], wMp[10], wKp[10];
    size_t whOff[10];
    {
        size_t off = 0;
        int cinAdj[4] = {20, 64, 128, 256};
        for (int i = 0; i < 4; ++i) {
            wsrc[i] = w_adj[i];
            wM[i] = adjCo[i];      wK[i] = 2 * cinAdj[i];
            wMp[i] = rup(wM[i], 64); wKp[i] = rup(wK[i], 32);
            whOff[i] = off;  off += (size_t)wMp[i] * wKp[i];
        }
        int cinConv[6] = {32, 64, 128, 256, 512, 1024};
        for (int i = 0; i < 6; ++i) {
            wsrc[4 + i] = w_conv[i];
            wM[4 + i] = convCo[i];        wK[4 + i] = 4 * cinConv[i];
            wMp[4 + i] = rup(wM[4 + i], 64); wKp[4 + i] = rup(wK[4 + i], 32);
            whOff[4 + i] = off;  off += (size_t)wMp[4 + i] * wKp[4 + i];
        }
        for (int i = 0; i < 10; ++i) {
            int n = wMp[i] * wKp[i];
            wcvt_kernel<<<(n + 255) / 256, 256, 0, stream>>>(
                wsrc[i], whBase + whOff[i], wM[i], wK[i], wMp[i], wKp[i]);
        }
    }

    const float* cur = seq;
    float* convOut = bC1;
    int T = 65536;

    for (int i = 0; i < 6; ++i) {
        int Cout = convCo[i];
        if (i < 4) {
            int L = Ls[i], E = Es[i];
            int Ca = adjCo[i];
            int Nadj = D * E;
            int wi = i;

            int nz = Ca * T;
            zero_kernel<<<(nz + 255) / 256, 256, 0, stream>>>(bAdj, nz);
            dim3 gA((unsigned)(Nadj / 128), (unsigned)(wMp[wi] / 64));
            wmma_stage_kernel<0><<<gA, 128, 0, stream>>>(
                cur, whBase + whOff[wi], adj_idx[i], bAdj,
                wM[wi], wK[wi], wKp[wi], wMp[wi], T,
                ilog2i(E), ilog2i(L), 0, 0);

            int wc = 4 + i;
            int Nc = T / 2;
            dim3 gC((unsigned)(Nc / 128), (unsigned)(wMp[wc] / 64));
            wmma_stage_kernel<1><<<gC, 128, 0, stream>>>(
                bAdj, whBase + whOff[wc], nullptr, convOut,
                wM[wc], wK[wc], wKp[wc], wMp[wc], T,
                0, 0, ilog2i(L / 2), ilog2i(L));
        } else {
            int wc = 4 + i;
            int Nc = T / 2;
            dim3 gC((unsigned)(Nc / 128), (unsigned)(wMp[wc] / 64));
            wmma_stage_kernel<1><<<gC, 128, 0, stream>>>(
                cur, whBase + whOff[wc], nullptr, convOut,
                wM[wc], wK[wc], wKp[wc], wMp[wc], T,
                0, 0, ilog2i(T / 2), ilog2i(T));
        }

        norm_lrelu_kernel<<<Cout, 256, 0, stream>>>(convOut, T / 2, (i >= 1) ? 1 : 0);

        cur = convOut;
        convOut = (convOut == bC1) ? bC2 : bC1;
        T /= 2;
    }

    final_kernel<<<(1024 + 255) / 256, 256, 0, stream>>>(cur, w_final, out, 2048, 1024);
}
